// WeiMultiHeadSelfAttention_5746666242530
// MI455X (gfx1250) — matrix operand order repaired and checked by reading, not compiled, not run on hardware
//
#include <hip/hip_runtime.h>
#include <hip/hip_bf16.h>

typedef _Float16 f16;
typedef _Float16 v16h __attribute__((ext_vector_type(16)));
typedef _Float16 v8h  __attribute__((ext_vector_type(8)));
typedef float    v8f  __attribute__((ext_vector_type(8)));

#define D_MODEL   1024
#define NUM_HEADS 16
#define D_HEAD    64
#define BATCH     4
#define SEQ       2048
#define HD        (NUM_HEADS * D_HEAD)   // 1024
#define MROWS     (BATCH * SEQ)          // 8192

union V16 { v16h v; v8h h[2]; };

// ---------------------------------------------------------------------------
// fp32 -> f16 conversion, 8 elements per thread (all sizes are multiples of 8)
// ---------------------------------------------------------------------------
__global__ __launch_bounds__(256)
void f32_to_f16_kernel(const float* __restrict__ in, f16* __restrict__ out, int n8) {
  int i = blockIdx.x * blockDim.x + threadIdx.x;
  if (i >= n8) return;
  const float4* p = (const float4*)(in) + (size_t)i * 2;
  float4 a = p[0], b = p[1];
  v8h o;
  o[0] = (f16)a.x; o[1] = (f16)a.y; o[2] = (f16)a.z; o[3] = (f16)a.w;
  o[4] = (f16)b.x; o[5] = (f16)b.y; o[6] = (f16)b.z; o[7] = (f16)b.w;
  *(v8h*)(out + (size_t)i * 8) = o;
}

// ---------------------------------------------------------------------------
// C[M,N] = A[M,K] * W[N,K]^T  via v_wmma_f32_16x16x32_f16.
// Block: 8 waves -> 128(M) x 64(N) tile; wave tile 16(M) x 64(N).
// W tile (64x32 f16 = 4KB) is staged block-cooperatively into LDS with
// global_load_async_to_lds_b128 (1 x 16B chunk per thread), double-buffered,
// synchronized with s_wait_asynccnt + barrier. B fragments then come from
// ds_load_b128 instead of 8x-redundant global loads.
// mode 0: store f16 row-major [M,N]
// mode 1: store f16 transposed per head: Vt[b][h][dh][SEQ]  (M=b*SEQ+s, N=h*64+dh)
// mode 2: store f32 row-major [M,N]
// ---------------------------------------------------------------------------
__global__ __launch_bounds__(256)
void gemm_xwT_wmma(const f16* __restrict__ A, const f16* __restrict__ W,
                   void* __restrict__ Cout, int M, int N, int K, int mode) {
  __shared__ f16 LdsB[2][64 * 32];   // 2 x 4KB double buffer

  const int tid  = threadIdx.x;
  const int lane = tid & 31;
  const int wave = tid >> 5;
  const int half = lane >> 4;
  const int l16  = lane & 15;
  const int nBase = blockIdx.x * 64;
  const int mBase = blockIdx.y * 128 + wave * 16;

  // cooperative fill: thread -> one 16B chunk of the 64(row) x 32(k) tile
  const int fRow = tid >> 2;          // 0..63  (n_local)
  const int fCol = (tid & 3) * 8;     // 0,8,16,24 (k_local)
  const f16* wSrc = W + (size_t)(nBase + fRow) * K + fCol;

  auto fill = [&](int buf, int k0) {
    unsigned lds = (unsigned)(unsigned long long)(const void*)&LdsB[buf][fRow * 32 + fCol];
    unsigned long long ga = (unsigned long long)(const void*)(wSrc + k0);
    asm volatile("global_load_async_to_lds_b128 %0, %1, off"
                 :: "v"(lds), "v"(ga) : "memory");
  };

  v8f acc[4] = {};
  const f16* aRow = A + (size_t)(mBase + l16) * K;

  const int L = K / 32;
  fill(0, 0);
  for (int i = 0; i < L; ++i) {
    asm volatile("s_wait_asynccnt 0x0" ::: "memory");   // own fill of cur done
    __syncthreads();                                    // everyone's fill done; prev reads done
    if (i + 1 < L) fill((i + 1) & 1, (i + 1) * 32);     // overlap next fill with compute

    const int k0 = i * 32;
    V16 ua;
    ua.h[0] = *(const v8h*)(aRow + k0 + half * 8);        // K = half*8 + j
    ua.h[1] = *(const v8h*)(aRow + k0 + 16 + half * 8);   // K = 16 + half*8 + j

    const f16* bTile = &LdsB[i & 1][0];
#pragma unroll
    for (int nt = 0; nt < 4; ++nt) {
      V16 ub;                                             // K = half*8 + j, 16 + half*8 + j
      ub.h[0] = *(const v8h*)(bTile + (nt * 16 + l16) * 32 + half * 8);
      ub.h[1] = *(const v8h*)(bTile + (nt * 16 + l16) * 32 + 16 + half * 8);
      acc[nt] = __builtin_amdgcn_wmma_f32_16x16x32_f16(
          false, ua.v, false, ub.v, (short)0, acc[nt], false, false);
    }
  }

  if (mode == 0) {
    f16* C = (f16*)Cout;
#pragma unroll
    for (int nt = 0; nt < 4; ++nt)
#pragma unroll
      for (int v = 0; v < 8; ++v)
        C[(size_t)(mBase + v + 8 * half) * N + nBase + nt * 16 + l16] = (f16)acc[nt][v];
  } else if (mode == 1) {
    f16* Vt = (f16*)Cout;
    int b = mBase / SEQ;
    int sBase = (mBase % SEQ) + 8 * half;   // 8 consecutive s per lane -> one b128 store
#pragma unroll
    for (int nt = 0; nt < 4; ++nt) {
      int n = nBase + nt * 16 + l16;
      int h = n >> 6, dh = n & 63;
      v8h pack;
#pragma unroll
      for (int v = 0; v < 8; ++v) pack[v] = (f16)acc[nt][v];
      *(v8h*)(Vt + (((size_t)b * NUM_HEADS + h) * D_HEAD + dh) * SEQ + sBase) = pack;
    }
  } else {
    float* C = (float*)Cout;
#pragma unroll
    for (int nt = 0; nt < 4; ++nt)
#pragma unroll
      for (int v = 0; v < 8; ++v)
        C[(size_t)(mBase + v + 8 * half) * N + nBase + nt * 16 + l16] = acc[nt][v];
  }
}

// ---------------------------------------------------------------------------
// Causal flash attention. One wave owns 16 query rows; 32 keys per iteration.
// Q,K layout: [B,S,HD] f16.  V layout: Vt[b][h][dh][SEQ] f16.  O: [B,S,HD] f16.
// Grid: B*H*(S/128) blocks of 256 threads (8 waves).
// ---------------------------------------------------------------------------
__global__ __launch_bounds__(256)
void attn_wmma(const f16* __restrict__ Q, const f16* __restrict__ Km,
               const f16* __restrict__ Vt, f16* __restrict__ O) {
  __shared__ f16 Plds[8][16 * 32];   // per-wave 16x32 P staging tile
  const int lane = threadIdx.x & 31;
  const int wave = threadIdx.x >> 5;
  const int half = lane >> 4;
  const int l16  = lane & 15;

  const int bid = blockIdx.x;
  const int qb  = bid & 15;          // S/128 = 16 q-blocks
  const int h   = (bid >> 4) & 15;
  const int b   = bid >> 8;
  const int q0  = qb * 128 + wave * 16;

  // Q fragments for this wave's 16 rows (Dh=64 -> two K=32 fragments), kept live
  const f16* qRow = Q + (size_t)(b * SEQ + q0 + l16) * HD + h * D_HEAD;
  V16 uq0, uq1;
  uq0.h[0] = *(const v8h*)(qRow + half * 8);
  uq0.h[1] = *(const v8h*)(qRow + 16 + half * 8);
  uq1.h[0] = *(const v8h*)(qRow + 32 + half * 8);
  uq1.h[1] = *(const v8h*)(qRow + 48 + half * 8);

  v8f o[4] = {};
  float mprev[8], lsum[8];
#pragma unroll
  for (int v = 0; v < 8; ++v) { mprev[v] = -1e30f; lsum[v] = 0.0f; }

  const f16* kBase = Km + (size_t)b * SEQ * HD + h * D_HEAD;
  const f16* vBase = Vt + ((size_t)b * NUM_HEADS + h) * D_HEAD * SEQ;
  f16* pW = &Plds[wave][0];

  for (int k0 = 0; k0 <= q0 + 15; k0 += 32) {
    // prefetch next key tile rows + next V segment while computing this one
    if (k0 + 32 < SEQ) {
      __builtin_prefetch(kBase + (size_t)(k0 + 32 + l16) * HD, 0, 3);
      __builtin_prefetch(vBase + (size_t)l16 * SEQ + k0 + 32, 0, 3);
    }

    // ---- S = Q * K^T for two 16-key sub-tiles ----
    v8f s0 = {}, s1 = {};
    {
      const f16* kRow = kBase + (size_t)(k0 + l16) * HD;
      V16 ub0, ub1;
      ub0.h[0] = *(const v8h*)(kRow + half * 8);
      ub0.h[1] = *(const v8h*)(kRow + 16 + half * 8);
      ub1.h[0] = *(const v8h*)(kRow + 32 + half * 8);
      ub1.h[1] = *(const v8h*)(kRow + 48 + half * 8);
      s0 = __builtin_amdgcn_wmma_f32_16x16x32_f16(false, uq0.v, false, ub0.v, (short)0, s0, false, false);
      s0 = __builtin_amdgcn_wmma_f32_16x16x32_f16(false, uq1.v, false, ub1.v, (short)0, s0, false, false);
    }
    {
      const f16* kRow = kBase + (size_t)(k0 + 16 + l16) * HD;
      V16 ub0, ub1;
      ub0.h[0] = *(const v8h*)(kRow + half * 8);
      ub0.h[1] = *(const v8h*)(kRow + 16 + half * 8);
      ub1.h[0] = *(const v8h*)(kRow + 32 + half * 8);
      ub1.h[1] = *(const v8h*)(kRow + 48 + half * 8);
      s1 = __builtin_amdgcn_wmma_f32_16x16x32_f16(false, uq0.v, false, ub0.v, (short)0, s1, false, false);
      s1 = __builtin_amdgcn_wmma_f32_16x16x32_f16(false, uq1.v, false, ub1.v, (short)0, s1, false, false);
    }

    // ---- online softmax over these 32 keys ----
#pragma unroll
    for (int v = 0; v < 8; ++v) {
      const int q   = q0 + v + 8 * half;       // global query row of this element
      const int kc0 = k0 + l16;                // global key col, tile 0
      const int kc1 = k0 + 16 + l16;           // global key col, tile 1
      float x0 = s0[v] * 0.125f;               // 1/sqrt(64)
      float x1 = s1[v] * 0.125f;
      if (kc0 > q) x0 = -1e30f;
      if (kc1 > q) x1 = -1e30f;
      float mx = fmaxf(x0, x1);
#pragma unroll
      for (int off = 8; off >= 1; off >>= 1) mx = fmaxf(mx, __shfl_xor(mx, off, 16));
      const float mnew = fmaxf(mprev[v], mx);
      const float p0 = __expf(x0 - mnew);
      const float p1 = __expf(x1 - mnew);
      float rs = p0 + p1;
#pragma unroll
      for (int off = 8; off >= 1; off >>= 1) rs += __shfl_xor(rs, off, 16);
      const float alpha = __expf(mprev[v] - mnew);
      lsum[v] = lsum[v] * alpha + rs;
      mprev[v] = mnew;
      o[0][v] *= alpha; o[1][v] *= alpha; o[2][v] *= alpha; o[3][v] *= alpha;
      const int row = v + 8 * half;
      pW[row * 32 + l16]      = (f16)p0;
      pW[row * 32 + 16 + l16] = (f16)p1;
    }
    // same-wave LDS RAW: wait DS writes, lanes are lockstep within the wave
    asm volatile("s_wait_dscnt 0x0" ::: "memory");

    // ---- re-load P in A-fragment layout ----
    V16 ap;
    const f16* pRow = pW + l16 * 32;
    ap.h[0] = *(const v8h*)(pRow + half * 8);
    ap.h[1] = *(const v8h*)(pRow + 16 + half * 8);

    // ---- O += P * V (Vt rows are contiguous in s) ----
#pragma unroll
    for (int nt = 0; nt < 4; ++nt) {
      const f16* vRow = vBase + (size_t)(nt * 16 + l16) * SEQ + k0 + half * 8;
      V16 ub;
      ub.h[0] = *(const v8h*)(vRow);
      ub.h[1] = *(const v8h*)(vRow + 16);
      o[nt] = __builtin_amdgcn_wmma_f32_16x16x32_f16(false, ap.v, false, ub.v, (short)0, o[nt], false, false);
    }
  }

  // ---- finalize: O /= l, store f16 [B,S,HD] ----
  float inv[8];
#pragma unroll
  for (int v = 0; v < 8; ++v) inv[v] = 1.0f / lsum[v];
#pragma unroll
  for (int nt = 0; nt < 4; ++nt)
#pragma unroll
    for (int v = 0; v < 8; ++v)
      O[(size_t)(b * SEQ + q0 + v + 8 * half) * HD + h * D_HEAD + nt * 16 + l16] =
          (f16)(o[nt][v] * inv[v]);
}

// ---------------------------------------------------------------------------
extern "C" void kernel_launch(void* const* d_in, const int* in_sizes, int n_in,
                              void* d_out, int out_size, void* d_ws, size_t ws_size,
                              hipStream_t stream) {
  (void)in_sizes; (void)n_in; (void)out_size; (void)ws_size;
  const float* X  = (const float*)d_in[0];
  const float* wq = (const float*)d_in[1];
  const float* wk = (const float*)d_in[2];
  const float* wv = (const float*)d_in[3];
  const float* wo = (const float*)d_in[4];

  // f16 workspace layout
  f16* ws = (f16*)d_ws;
  size_t off = 0;
  f16* Xh  = ws + off; off += (size_t)MROWS * D_MODEL;   // 8M
  f16* Wqh = ws + off; off += (size_t)HD * D_MODEL;      // 1M
  f16* Wkh = ws + off; off += (size_t)HD * D_MODEL;
  f16* Wvh = ws + off; off += (size_t)HD * D_MODEL;
  f16* Woh = ws + off; off += (size_t)D_MODEL * HD;
  f16* Qh  = ws + off; off += (size_t)MROWS * HD;        // 8M
  f16* Kh  = ws + off; off += (size_t)MROWS * HD;        // 8M
  f16* VtH = ws + off; off += (size_t)MROWS * HD;        // 8M, [b][h][dh][SEQ]
  f16* Oh  = Xh;  // alias: X no longer needed once attention runs

  auto cvt = [&](const float* src, f16* dst, size_t n) {
    int n8 = (int)(n / 8);
    f32_to_f16_kernel<<<(n8 + 255) / 256, 256, 0, stream>>>(src, dst, n8);
  };
  cvt(X,  Xh,  (size_t)MROWS * D_MODEL);
  cvt(wq, Wqh, (size_t)HD * D_MODEL);
  cvt(wk, Wkh, (size_t)HD * D_MODEL);
  cvt(wv, Wvh, (size_t)HD * D_MODEL);
  cvt(wo, Woh, (size_t)D_MODEL * HD);

  // QKV projections: M=8192, N=1024, K=1024
  dim3 gProj(HD / 64, MROWS / 128);
  gemm_xwT_wmma<<<gProj, 256, 0, stream>>>(Xh, Wqh, Qh,  MROWS, HD, D_MODEL, 0);
  gemm_xwT_wmma<<<gProj, 256, 0, stream>>>(Xh, Wkh, Kh,  MROWS, HD, D_MODEL, 0);
  gemm_xwT_wmma<<<gProj, 256, 0, stream>>>(Xh, Wvh, VtH, MROWS, HD, D_MODEL, 1);

  // attention: B*H*(S/128) blocks
  attn_wmma<<<BATCH * NUM_HEADS * (SEQ / 128), 256, 0, stream>>>(Qh, Kh, VtH, Oh);

  // output projection -> fp32 d_out
  gemm_xwT_wmma<<<dim3(D_MODEL / 64, MROWS / 128), 256, 0, stream>>>(
      Oh, Woh, d_out, MROWS, D_MODEL, HD, 2);
}
